// DiffTimeLanguageModel_58119497449794
// MI455X (gfx1250) — matrix operand order repaired and checked by reading, not compiled, not run on hardware
//
#include <hip/hip_runtime.h>
#include <hip/hip_bf16.h>
#include <math.h>

#define NTOKEN 32000
#define NWE    512
#define H_RNN  1024
#define SEQ    128
#define BATCH  32
#define NTOK   (SEQ * BATCH)   // 4096
#define NTS_F  100.0f

typedef __attribute__((ext_vector_type(16))) __bf16 v16bf;
typedef __attribute__((ext_vector_type(8)))  float  v8f;

// ---------------------------------------------------------------------------
// WMMA fragment loaders from bf16 memory (CDNA5 wave32 layouts).
// A 16x32 bf16: lane m = lane&15, half = lane>>4; elem e: k = half*8+e (e<8),
//               16+half*8+(e-8) (e>=8)  -> two 16-byte runs.
// B 32x16 bf16: lane n = lane&15; elem e: same k order as A -> two 16-byte runs.
// All row bases are >=16B aligned (lda/ldw multiples of 64, kb multiple of 32).
// ---------------------------------------------------------------------------
__device__ __forceinline__ v16bf load_a_frag16(const __bf16* __restrict__ row,
                                               int kb, int half) {
  const __bf16* ap = row + kb + half * 8;
  union { uint4 q[2]; v16bf v; } u;
  u.q[0] = *(const uint4*)(ap);        // k = half*8 + 0..7
  u.q[1] = *(const uint4*)(ap + 16);   // k = 16 + half*8 + 0..7
  return u.v;
}

__device__ __forceinline__ v16bf load_b_frag16(const __bf16* __restrict__ row,
                                               int kb, int half) {
  const __bf16* bp = row + kb + half * 8;
  union { uint4 q[2]; v16bf v; } u;
  u.q[0] = *(const uint4*)(bp);
  u.q[1] = *(const uint4*)(bp + 16);
  return u.v;
}

// ---------------------------------------------------------------------------
// Generic batched GEMM, bf16 operands, f32 accumulate:
//   C[b][m][n] (f32 and/or bf16) = sum_k A[b][m][k] * W[b][n][k] + bias(n)
// One wave computes a 16x64 tile: 4 WMMA accumulators share one A fragment
// (4x A-load reuse). Optional A-row gather (embedding lookup).
// ---------------------------------------------------------------------------
__global__ void gemm_wmma_kernel(
    const __bf16* __restrict__ A, long lda, long a_bstride,
    const __bf16* __restrict__ W, long ldw, long w_bstride,
    const float* __restrict__ bias1, long b1_bstride,
    const float* __restrict__ bias2,
    float* __restrict__ C, __bf16* __restrict__ Ch, long ldc, long c_bstride,
    int M, int N, int K,
    const int* __restrict__ a_gather, int a_gstride)
{
  const int lane = threadIdx.x & 31;
  const int wave = (int)((blockIdx.x * (long)blockDim.x + threadIdx.x) >> 5);
  const int nt64 = N >> 6;
  const int m_tiles = M >> 4;
  if (wave >= nt64 * m_tiles) return;   // uniform whole-wave exit: EXEC stays full
  const int tm = wave / nt64;
  const int tn = wave - tm * nt64;
  const int batch = blockIdx.y;
  const int half = lane >> 4;
  const int l15  = lane & 15;
  const int m_base = tm << 4;
  const int n_base = tn << 6;

  const __bf16* Ab = A + (long)batch * a_bstride;
  const __bf16* Wb = W + (long)batch * w_bstride;

  long arow = (long)(m_base + l15);
  if (a_gather) arow = (long)a_gather[arow * a_gstride + batch];
  const __bf16* arow_ptr = Ab + arow * lda;
  const __bf16* w0 = Wb + (long)(n_base + l15) * ldw;
  const __bf16* w1 = w0 + 16 * ldw;
  const __bf16* w2 = w0 + 32 * ldw;
  const __bf16* w3 = w0 + 48 * ldw;

  v8f acc[4] = {v8f{}, v8f{}, v8f{}, v8f{}};
  for (int kb = 0; kb < K; kb += 32) {
    const v16bf a = load_a_frag16(arow_ptr, kb, half);
    acc[0] = __builtin_amdgcn_wmma_f32_16x16x32_bf16(
        false, a, false, load_b_frag16(w0, kb, half), (short)0, acc[0], false, false);
    acc[1] = __builtin_amdgcn_wmma_f32_16x16x32_bf16(
        false, a, false, load_b_frag16(w1, kb, half), (short)0, acc[1], false, false);
    acc[2] = __builtin_amdgcn_wmma_f32_16x16x32_bf16(
        false, a, false, load_b_frag16(w2, kb, half), (short)0, acc[2], false, false);
    acc[3] = __builtin_amdgcn_wmma_f32_16x16x32_bf16(
        false, a, false, load_b_frag16(w3, kb, half), (short)0, acc[3], false, false);
    // gfx1250 prefetch of next K-chunk of the streamed operands
    __builtin_prefetch(w0 + kb + 32, 0, 1);
    __builtin_prefetch(arow_ptr + kb + 32, 0, 1);
  }

  float*  Cb  = C  ? (C  + (long)batch * c_bstride) : nullptr;
  __bf16* Chb = Ch ? (Ch + (long)batch * c_bstride) : nullptr;
  const int row0 = m_base + half * 8;   // C/D layout: VGPR r -> M = half*8+r
#pragma unroll
  for (int j = 0; j < 4; ++j) {
    const int col = n_base + j * 16 + l15;
    float bv = 0.f;
    if (bias1) bv += bias1[(long)batch * b1_bstride + col];
    if (bias2) bv += bias2[col];
#pragma unroll
    for (int r = 0; r < 8; ++r) {
      const float v = acc[j][r] + bv;
      const long idx = (long)(row0 + r) * ldc + col;
      if (Cb)  Cb[idx]  = v;
      if (Chb) Chb[idx] = (__bf16)v;
    }
  }
}

// ---------------------------------------------------------------------------
// Fused LSTM step: gates = xp[t] + h @ Whh^T (4 gate tiles share the A
// fragment), then elementwise update. h state lives entirely in bf16 (it is
// only ever consumed as a WMMA A-operand); c stays f32. h ping-pongs through
// the bf16 ys buffer (h_in = ys[t-1], h_out = ys[t]) to avoid RAW races.
// 128 waves (2 m-tiles x 64 n-tiles), H = 1024.
// ---------------------------------------------------------------------------
__global__ void lstm_step_kernel(
    const float*  __restrict__ xp_t,   // [32, 4H] f32 (bih+bhh folded in)
    const __bf16* __restrict__ Whh,    // [4H, H] bf16
    const __bf16* __restrict__ h_in,   // [32, H] bf16
    __bf16* __restrict__ h_out,        // [32, H] bf16 (= ys[t])
    float* __restrict__ c_state)       // [32, H] f32
{
  const int H = H_RNN;
  const int lane = threadIdx.x & 31;
  const int wave = (int)((blockIdx.x * (long)blockDim.x + threadIdx.x) >> 5);
  const int n_tiles = H >> 4;               // 64
  if (wave >= 2 * n_tiles) return;
  const int tm = wave / n_tiles;
  const int tn = wave - tm * n_tiles;
  const int half = lane >> 4;
  const int l15  = lane & 15;
  const int m_base = tm << 4;
  const int n_base = tn << 4;
  const int ncol = n_base + l15;

  const __bf16* arow = h_in + (long)(m_base + l15) * H;
  const __bf16* wi = Whh + (long)(0 * H + ncol) * H;
  const __bf16* wf = Whh + (long)(1 * H + ncol) * H;
  const __bf16* wg = Whh + (long)(2 * H + ncol) * H;
  const __bf16* wo = Whh + (long)(3 * H + ncol) * H;

  v8f ai = {}, af = {}, ag = {}, ao = {};
  for (int kb = 0; kb < H; kb += 32) {
    const v16bf a = load_a_frag16(arow, kb, half);
    ai = __builtin_amdgcn_wmma_f32_16x16x32_bf16(
        false, a, false, load_b_frag16(wi, kb, half), (short)0, ai, false, false);
    af = __builtin_amdgcn_wmma_f32_16x16x32_bf16(
        false, a, false, load_b_frag16(wf, kb, half), (short)0, af, false, false);
    ag = __builtin_amdgcn_wmma_f32_16x16x32_bf16(
        false, a, false, load_b_frag16(wg, kb, half), (short)0, ag, false, false);
    ao = __builtin_amdgcn_wmma_f32_16x16x32_bf16(
        false, a, false, load_b_frag16(wo, kb, half), (short)0, ao, false, false);
  }

  const int row0 = m_base + half * 8;
#pragma unroll
  for (int r = 0; r < 8; ++r) {
    const int m = row0 + r;
    const long x4 = (long)m * (4 * H);
    float gi = ai[r] + xp_t[x4 + ncol];
    float gf = af[r] + xp_t[x4 + H + ncol];
    float gg = ag[r] + xp_t[x4 + 2 * H + ncol];
    float go = ao[r] + xp_t[x4 + 3 * H + ncol];
    gi = 1.f / (1.f + __expf(-gi));
    gf = 1.f / (1.f + __expf(-gf));
    gg = tanhf(gg);
    go = 1.f / (1.f + __expf(-go));
    const long idx = (long)m * H + ncol;
    const float cc = gf * c_state[idx] + gi * gg;   // c owned by exactly one lane
    c_state[idx] = cc;
    h_out[idx] = (__bf16)(go * tanhf(cc));
  }
}

// ---------------------------------------------------------------------------
// f32 -> bf16 conversion (one pass per weight tensor; n must be /4)
// ---------------------------------------------------------------------------
__global__ void cvt_bf16_kernel(const float* __restrict__ src,
                                __bf16* __restrict__ dst, long n4) {
  const long i = (long)blockIdx.x * blockDim.x + threadIdx.x;
  if (i >= n4) return;
  const float4 f = ((const float4*)src)[i];
  union { __bf16 h[4]; uint2 u; } o;
  o.h[0] = (__bf16)f.x; o.h[1] = (__bf16)f.y;
  o.h[2] = (__bf16)f.z; o.h[3] = (__bf16)f.w;
  ((uint2*)dst)[i] = o.u;
}

// ---------------------------------------------------------------------------
// Front-end smalls
// ---------------------------------------------------------------------------
__global__ void frontend_kernel(const int* __restrict__ timestep,
                                const float* __restrict__ tc_W1,
                                const float* __restrict__ tc_b1,
                                const float* __restrict__ tc_W2,
                                const float* __restrict__ tc_b2,
                                const float* __restrict__ trans_b,
                                float* __restrict__ ht,    // [32,64]
                                float* __restrict__ tbv)   // [32,64] trans_b . ht
{
  const int b = blockIdx.x;     // 32
  const int i = threadIdx.x;    // 64
  __shared__ float h1s[64];
  __shared__ float hts[64];
  const float ts = (float)timestep[b] / NTS_F;
  h1s[i] = tanhf(ts * tc_W1[i] + tc_b1[i]);
  __syncthreads();
  float acc = tc_b2[i];
  for (int j = 0; j < 64; ++j) acc += tc_W2[i * 64 + j] * h1s[j];
  const float h = tanhf(acc);
  hts[i] = h;
  ht[b * 64 + i] = h;
  __syncthreads();
  float tb = 0.f;
  for (int j = 0; j < 64; ++j) tb += trans_b[i * 64 + j] * hts[j];
  tbv[b * 64 + i] = tb;
}

// WbT[b][k][i] = sum_j trans_W[(i*64+j)*512 + k] * ht[b][j]   (bf16 out: it is
// consumed only as the W-operand of the Cb GEMM)
__global__ void make_wbt_kernel(const float* __restrict__ trans_W,
                                const float* __restrict__ ht,
                                __bf16* __restrict__ WbT)   // [32,512,64] bf16
{
  const int k = blockIdx.x;     // 512
  const int b = blockIdx.y;     // 32
  const int i = threadIdx.x;    // 64
  __shared__ float hts[64];
  hts[i] = ht[b * 64 + i];
  __syncthreads();
  float acc = 0.f;
  for (int j = 0; j < 64; ++j)
    acc += trans_W[((long)(i * 64 + j)) * NWE + k] * hts[j];
  WbT[((long)b * NWE + k) * 64 + i] = (__bf16)acc;
}

// ebias[b][o] = dw_b[o] + sum_i dw_W[o][i] * tbv[b][i]   (trans_b fold)
__global__ void make_ebias_kernel(const float* __restrict__ dw_W,
                                  const float* __restrict__ dw_b,
                                  const float* __restrict__ tbv,
                                  float* __restrict__ ebias) // [32,512]
{
  const int o = blockIdx.x * blockDim.x + threadIdx.x;
  const int b = blockIdx.y;
  if (o >= NWE) return;
  float acc = dw_b[o];
  for (int i = 0; i < 64; ++i) acc += dw_W[o * 64 + i] * tbv[b * 64 + i];
  ebias[b * NWE + o] = acc;
}

__global__ void zero_kernel(float* __restrict__ p, long n) {
  const long i = (long)blockIdx.x * blockDim.x + threadIdx.x;
  if (i < n) p[i] = 0.f;
}

// ---------------------------------------------------------------------------
extern "C" void kernel_launch(void* const* d_in, const int* in_sizes, int n_in,
                              void* d_out, int out_size, void* d_ws, size_t ws_size,
                              hipStream_t stream) {
  (void)in_sizes; (void)n_in; (void)out_size; (void)ws_size;
  const int*   text     = (const int*)  d_in[0];
  const int*   timestep = (const int*)  d_in[1];
  const float* U_weight = (const float*)d_in[2];
  const float* trans_W  = (const float*)d_in[3];
  const float* trans_b  = (const float*)d_in[4];
  const float* tc_W1    = (const float*)d_in[5];
  const float* tc_b1    = (const float*)d_in[6];
  const float* tc_W2    = (const float*)d_in[7];
  const float* tc_b2    = (const float*)d_in[8];
  const float* dw_W     = (const float*)d_in[9];
  const float* dw_b     = (const float*)d_in[10];
  const float* Wih0     = (const float*)d_in[11];
  const float* Whh0     = (const float*)d_in[12];
  const float* bih0     = (const float*)d_in[13];
  const float* bhh0     = (const float*)d_in[14];
  const float* Wih1     = (const float*)d_in[15];
  const float* Whh1     = (const float*)d_in[16];
  const float* bih1     = (const float*)d_in[17];
  const float* bhh1     = (const float*)d_in[18];
  const float* dec_W    = (const float*)d_in[19];
  const float* dec_b    = (const float*)d_in[20];
  float* logits = (float*)d_out;

  char* ws = (char*)d_ws;
  size_t off = 0;
  auto allocB = [&](size_t bytes) {
    char* p = ws + off;
    off += (bytes + 255) & ~(size_t)255;
    return p;
  };
  // f32 scratch
  float* htB   = (float*)allocB((size_t)BATCH * 64 * 4);
  float* tbvB  = (float*)allocB((size_t)BATCH * 64 * 4);
  float* ebias = (float*)allocB((size_t)BATCH * NWE * 4);
  float* xpB   = (float*)allocB((size_t)NTOK * 4 * H_RNN * 4);   // 64 MB, reused
  float* cB    = (float*)allocB((size_t)BATCH * H_RNN * 4);
  // bf16 scratch (operands of WMMA)
  __bf16* WbT   = (__bf16*)allocB((size_t)BATCH * NWE * 64 * 2);
  __bf16* CbB   = (__bf16*)allocB((size_t)BATCH * NWE * NWE * 2);   // 16.8 MB
  __bf16* embB  = (__bf16*)allocB((size_t)NTOK * NWE * 2);          // 4.2 MB
  __bf16* ys0   = (__bf16*)allocB((size_t)NTOK * H_RNN * 2);        // 8.4 MB
  __bf16* ys1   = (__bf16*)allocB((size_t)NTOK * H_RNN * 2);        // 8.4 MB
  __bf16* h0z   = (__bf16*)allocB((size_t)BATCH * H_RNN * 2);
  // bf16 weight copies (converted once; dec 65.5 MB -> L2 resident)
  __bf16* Ubf   = (__bf16*)allocB((size_t)NTOKEN * NWE * 2);
  __bf16* dwWbf = (__bf16*)allocB((size_t)NWE * 64 * 2);
  __bf16* Wih0b = (__bf16*)allocB((size_t)4 * H_RNN * NWE * 2);
  __bf16* Whh0b = (__bf16*)allocB((size_t)4 * H_RNN * H_RNN * 2);
  __bf16* Wih1b = (__bf16*)allocB((size_t)4 * H_RNN * H_RNN * 2);
  __bf16* Whh1b = (__bf16*)allocB((size_t)4 * H_RNN * H_RNN * 2);
  __bf16* decWb = (__bf16*)allocB((size_t)NTOKEN * H_RNN * 2);

  auto cvt = [&](const float* s, __bf16* d, long n) {
    const long n4 = n / 4;
    cvt_bf16_kernel<<<dim3((int)((n4 + 255) / 256)), dim3(256), 0, stream>>>(s, d, n4);
  };
  auto launch_gemm = [&](const __bf16* A, long lda, long abs_,
                         const __bf16* W, long ldw, long wbs,
                         const float* b1, long b1bs, const float* b2,
                         float* C, __bf16* Ch, long ldc, long cbs,
                         int M, int N, int K,
                         const int* gath, int gstride, int nbatch) {
    const int waves  = (M / 16) * (N / 64);
    const int blocks = (waves * 32 + 255) / 256;
    gemm_wmma_kernel<<<dim3(blocks, nbatch), dim3(256), 0, stream>>>(
        A, lda, abs_, W, ldw, wbs, b1, b1bs, b2, C, Ch, ldc, cbs, M, N, K, gath, gstride);
  };

  // ---- one-time weight conversions (per call) ----
  cvt(U_weight, Ubf,   (long)NTOKEN * NWE);
  cvt(dw_W,     dwWbf, (long)NWE * 64);
  cvt(Wih0,     Wih0b, (long)4 * H_RNN * NWE);
  cvt(Whh0,     Whh0b, (long)4 * H_RNN * H_RNN);
  cvt(Wih1,     Wih1b, (long)4 * H_RNN * H_RNN);
  cvt(Whh1,     Whh1b, (long)4 * H_RNN * H_RNN);
  cvt(dec_W,    decWb, (long)NTOKEN * H_RNN);

  // zero h0 (bf16 zeros are bit-zero) and c
  {
    const long nf = (long)BATCH * H_RNN / 2;   // bf16 region as f32 words
    zero_kernel<<<dim3((int)((nf + 255) / 256)), dim3(256), 0, stream>>>((float*)h0z, nf);
  }

  // ---- front-end (algebraically refactored) ----
  frontend_kernel<<<dim3(BATCH), dim3(64), 0, stream>>>(
      timestep, tc_W1, tc_b1, tc_W2, tc_b2, trans_b, htB, tbvB);
  make_wbt_kernel<<<dim3(NWE, BATCH), dim3(64), 0, stream>>>(trans_W, htB, WbT);
  make_ebias_kernel<<<dim3(1, BATCH), dim3(512), 0, stream>>>(dw_W, dw_b, tbvB, ebias);

  // Cb[b] = dw_W @ Wb[b] : C[o][k2] = sum_i dw_W[o][i] * WbT[b][k2][i]  (bf16 out)
  launch_gemm(dwWbf, 64, 0, WbT, 64, (long)NWE * 64, nullptr, 0, nullptr,
              nullptr, CbB, NWE, (long)NWE * NWE, NWE, NWE, 64, nullptr, 0, BATCH);

  // emb[s][b][:] = Cb[b] @ U[text[s][b]] + ebias[b]   (A-row gather, bf16 out)
  launch_gemm(Ubf, NWE, 0, CbB, NWE, (long)NWE * NWE, ebias, NWE, nullptr,
              nullptr, embB, (long)BATCH * NWE, NWE, SEQ, NWE, NWE, text, BATCH, BATCH);

  // ---- LSTM layer 0 ----
  launch_gemm(embB, NWE, 0, Wih0b, NWE, 0, bih0, 0, bhh0,
              xpB, nullptr, 4 * H_RNN, 0, NTOK, 4 * H_RNN, NWE, nullptr, 0, 1);
  {
    const long n = (long)BATCH * H_RNN;
    zero_kernel<<<dim3((int)((n + 255) / 256)), dim3(256), 0, stream>>>(cB, n);
  }
  for (int t = 0; t < SEQ; ++t) {
    const __bf16* hin = (t == 0) ? h0z : (ys0 + (long)(t - 1) * BATCH * H_RNN);
    lstm_step_kernel<<<dim3(16), dim3(256), 0, stream>>>(
        xpB + (long)t * BATCH * 4 * H_RNN, Whh0b, hin,
        ys0 + (long)t * BATCH * H_RNN, cB);
  }

  // ---- LSTM layer 1 ----
  launch_gemm(ys0, H_RNN, 0, Wih1b, H_RNN, 0, bih1, 0, bhh1,
              xpB, nullptr, 4 * H_RNN, 0, NTOK, 4 * H_RNN, H_RNN, nullptr, 0, 1);
  {
    const long n = (long)BATCH * H_RNN;
    zero_kernel<<<dim3((int)((n + 255) / 256)), dim3(256), 0, stream>>>(cB, n);
  }
  for (int t = 0; t < SEQ; ++t) {
    const __bf16* hin = (t == 0) ? h0z : (ys1 + (long)(t - 1) * BATCH * H_RNN);
    lstm_step_kernel<<<dim3(16), dim3(256), 0, stream>>>(
        xpB + (long)t * BATCH * 4 * H_RNN, Whh1b, hin,
        ys1 + (long)t * BATCH * H_RNN, cB);
  }

  // ---- decoder: logits = ys1 @ dec_W^T + dec_b  (dominant, 268 GFLOP) ----
  launch_gemm(ys1, H_RNN, 0, decWb, H_RNN, 0, dec_b, 0, nullptr,
              logits, nullptr, NTOKEN, 0, NTOK, NTOKEN, H_RNN, nullptr, 0, 1);
}
